// PositionConditionerBlock_52845277609997
// MI455X (gfx1250) — matrix operand order repaired and checked by reading, not compiled, not run on hardware
//
#include <hip/hip_runtime.h>

typedef __attribute__((ext_vector_type(16))) _Float16 v16h;
typedef __attribute__((ext_vector_type(8)))  _Float16 v8h;
typedef __attribute__((ext_vector_type(8)))  float    v8f;
typedef __attribute__((ext_vector_type(4)))  float    f32x4;

#define WMMA_F32_F16(a,b,c) \
  __builtin_amdgcn_wmma_f32_16x16x32_f16(false,(a),false,(b),(short)0,(c),false,false)

#define SEQ   2048
#define NH24  24
#define CDIM  64
#define DDIM  256
#define KQLD  (NH24*CDIM)     // 1536, row stride of k/q buffers
#define VLD   (NH24*DDIM)     // 6144, row stride of val buffer
#define JT    32              // j-tile (WMMA K for the P@V GEMM)
#define VT_S  40              // LDS stride (halves) for transposed val tile
#define P_S   40              // LDS stride (halves) for P tile

// ---------------- fragment loaders (CDNA5 wave32 WMMA layouts) ----------------

// A-matrix 16x32 f16, row-major source, lda in elements (multiple of 8).
// lanes 0-15: M=lane, K={kb..kb+7, kb+16..kb+23} with kb=0; lanes 16-31: kb=8.
static __device__ __forceinline__ v16h frag_a_f16(const _Float16* A, int lda) {
  const int l   = threadIdx.x & 31;
  const int row = l & 15;
  const int kb  = (l >> 4) * 8;
  const _Float16* p = A + (size_t)row * lda + kb;
  union { v16h v; f32x4 f[2]; } u;
  u.f[0] = *(const f32x4*)(p);
  u.f[1] = *(const f32x4*)(p + 16);
  return u.v;
}

// B-matrix 32x16 f16 from row-major [N x K] source (i.e. B = srcT).
// lanes 0-15: N=lane, K={kb..kb+7, kb+16..kb+23} with kb=0; lanes 16-31: N=lane-16, kb=8.
static __device__ __forceinline__ v16h frag_b_f16(const _Float16* B, int ldb) {
  const int l  = threadIdx.x & 31;
  const int n  = l & 15;
  const int kb = (l >> 4) * 8;
  const _Float16* p = B + (size_t)n * ldb + kb;
  union { v16h v; f32x4 f[2]; } u;
  u.f[0] = *(const f32x4*)(p);
  u.f[1] = *(const f32x4*)(p + 16);
  return u.v;
}

static __device__ __forceinline__ v16h frag_a_f16_guard(const _Float16* A, int lda, int krem) {
  const int l   = threadIdx.x & 31;
  const int row = l & 15;
  const int kb  = (l >> 4) * 8;
  v16h r;
  #pragma unroll
  for (int i = 0; i < 8; ++i) {
    const int k0 = kb + i, k1 = kb + 16 + i;
    r[i]     = (k0 < krem) ? A[(size_t)row * lda + k0] : (_Float16)0.f;
    r[i + 8] = (k1 < krem) ? A[(size_t)row * lda + k1] : (_Float16)0.f;
  }
  return r;
}

static __device__ __forceinline__ v16h frag_b_f16_guard(const _Float16* B, int ldb, int krem) {
  const int l  = threadIdx.x & 31;
  const int n  = l & 15;
  const int kb = (l >> 4) * 8;
  v16h r;
  #pragma unroll
  for (int i = 0; i < 16; ++i) {
    const int k = kb + (i < 8 ? i : i + 8);
    r[i] = (k < krem) ? B[(size_t)n * ldb + k] : (_Float16)0.f;
  }
  return r;
}

// ---------------- elementwise fp32 -> f16 ----------------
__global__ __launch_bounds__(256)
void cvt_f32_to_f16(const float* __restrict__ s, _Float16* __restrict__ d, int n) {
  const int i = blockIdx.x * 256 + threadIdx.x;
  if (i < n) d[i] = (_Float16)s[i];
}

// ---------------- WMMA GEMM:  C(MxN) = A(MxK) * B(NxK)^T + bias ----------------
// One wave per 16x16 output tile. If kdst != nullptr, output columns are the
// packed [head(8) x (k|q)(2x64)] layout and get scattered into the concatenated
// k/q buffers of shape (S, 24, 64) at head offset head_off.
__global__ __launch_bounds__(32)
void wmma_proj_gemm(const _Float16* __restrict__ A, const _Float16* __restrict__ Bm,
                    const float* __restrict__ bias,
                    _Float16* __restrict__ Cplain, int ldc,
                    _Float16* __restrict__ kdst, _Float16* __restrict__ qdst, int head_off,
                    int K) {
  const int n0 = blockIdx.x * 16;
  const int m0 = blockIdx.y * 16;
  const _Float16* Ab = A  + (size_t)m0 * K;
  const _Float16* Bb = Bm + (size_t)n0 * K;

  v8f acc = {};
  int k = 0;
  for (; k + 32 <= K; k += 32) {
    v16h a = frag_a_f16(Ab + k, K);
    v16h b = frag_b_f16(Bb + k, K);
    acc = WMMA_F32_F16(a, b, acc);
  }
  if (k < K) {
    v16h a = frag_a_f16_guard(Ab + k, K, K - k);
    v16h b = frag_b_f16_guard(Bb + k, K, K - k);
    acc = WMMA_F32_F16(a, b, acc);
  }

  const int l   = threadIdx.x & 31;
  const int col = l & 15;
  const int rb  = (l >> 4) * 8;
  const float bv = bias ? bias[n0 + col] : 0.0f;

  #pragma unroll
  for (int r = 0; r < 8; ++r) {
    const int row = m0 + rb + r;
    const _Float16 hv = (_Float16)(acc[r] + bv);
    if (kdst) {
      const int cg = n0 + col;        // 0..1023
      const int h  = cg >> 7;         // head within source (0..7)
      const int c  = cg & 127;        // 0..127: [0,64)=k, [64,128)=q
      if (c < 64)
        kdst[(size_t)row * KQLD + (head_off + h) * CDIM + c] = hv;
      else
        qdst[(size_t)row * KQLD + (head_off + h) * CDIM + (c - 64)] = hv;
    } else {
      Cplain[(size_t)row * ldc + n0 + col] = hv;
    }
  }
}

// ---------------- fused flash attention over j, per (head, 64-row i-group) ----
// block = 128 threads = 4 waves; wave w owns i-tile rows [i0, i0+16).
// All waves share head h => val/q tiles of the j-loop are reused 4x.
__global__ __launch_bounds__(128)
void flash_attn_wmma(const _Float16* __restrict__ kbuf, const _Float16* __restrict__ qbuf,
                     const _Float16* __restrict__ val, float* __restrict__ out) {
  __shared__ _Float16 vt[DDIM * VT_S];     // transposed val tile: [d][j]  (20.0 KB)
  __shared__ _Float16 pbuf[4 * 16 * P_S];  // per-wave P tiles 16x32       ( 5.0 KB)

  const int h   = blockIdx.y;
  const int w   = threadIdx.x >> 5;
  const int l   = threadIdx.x & 31;
  const int ln  = l & 15;
  const int lh  = l >> 4;
  const int i0  = (blockIdx.x * 4 + w) * 16;

  // K fragments for this i-tile / head (A of the logits GEMM), C=64 -> 2 chunks
  const _Float16* kb = kbuf + (size_t)i0 * KQLD + h * CDIM;
  const v16h ak0 = frag_a_f16(kb, KQLD);
  const v16h ak1 = frag_a_f16(kb + 32, KQLD);

  v8f acc[16];
  #pragma unroll
  for (int t = 0; t < 16; ++t) { v8f z = {}; acc[t] = z; }
  float m8[8], s8[8];
  #pragma unroll
  for (int r = 0; r < 8; ++r) { m8[r] = -3.0e38f; s8[r] = 0.f; }

  const int jl = threadIdx.x >> 2;  // 0..31 : j row staged by this thread
  const int dq = threadIdx.x & 3;   // 0..3  : 64-wide d chunk

  for (int j0 = 0; j0 < SEQ; j0 += JT) {
    __syncthreads();  // previous iteration's LDS reads complete
    // ---- stage val tile transposed into LDS: vt[d][j - j0] ----
    {
      const _Float16* vrow = val + (size_t)(j0 + jl) * VLD + h * DDIM + dq * 64;
      #pragma unroll
      for (int c = 0; c < 64; c += 8) {
        const v8h u = *(const v8h*)(vrow + c);
        const int d = dq * 64 + c;
        #pragma unroll
        for (int e = 0; e < 8; ++e) vt[(d + e) * VT_S + jl] = u[e];
      }
    }
    if (j0 + JT < SEQ)  // warm L2 for next q tile
      __builtin_prefetch(qbuf + (size_t)(j0 + JT) * KQLD + h * CDIM, 0, 1);
    __syncthreads();

    // ---- logits tile (16 rows i) x (32 cols j), K = C = 64 ----
    v8f lg0 = {}, lg1 = {};
    {
      const _Float16* qb0 = qbuf + (size_t)j0 * KQLD + h * CDIM;
      lg0 = WMMA_F32_F16(ak0, frag_b_f16(qb0, KQLD), lg0);
      lg0 = WMMA_F32_F16(ak1, frag_b_f16(qb0 + 32, KQLD), lg0);
      const _Float16* qb1 = qbuf + (size_t)(j0 + 16) * KQLD + h * CDIM;
      lg1 = WMMA_F32_F16(ak0, frag_b_f16(qb1, KQLD), lg1);
      lg1 = WMMA_F32_F16(ak1, frag_b_f16(qb1 + 32, KQLD), lg1);
    }

    // ---- online softmax over j (columns) ----
    float p0[8], p1[8], sc[8];
    #pragma unroll
    for (int r = 0; r < 8; ++r) {
      const float a0 = lg0[r] * 0.35355339059327373f;  // 1/sqrt(H=8)
      const float a1 = lg1[r] * 0.35355339059327373f;
      float vm = fmaxf(a0, a1);
      #pragma unroll
      for (int msk = 1; msk < 16; msk <<= 1) vm = fmaxf(vm, __shfl_xor(vm, msk, 32));
      const float nm = fmaxf(m8[r], vm);
      sc[r] = __expf(m8[r] - nm);
      p0[r] = __expf(a0 - nm);
      p1[r] = __expf(a1 - nm);
      float rs = p0[r] + p1[r];
      #pragma unroll
      for (int msk = 1; msk < 16; msk <<= 1) rs += __shfl_xor(rs, msk, 32);
      s8[r] = s8[r] * sc[r] + rs;
      m8[r] = nm;
    }
    #pragma unroll
    for (int t = 0; t < 16; ++t)
      #pragma unroll
      for (int r = 0; r < 8; ++r) acc[t][r] *= sc[r];

    // ---- P (16x32) through LDS to re-fragment D-layout -> A-layout ----
    _Float16* pw = pbuf + w * 16 * P_S;
    #pragma unroll
    for (int r = 0; r < 8; ++r) {
      const int row = lh * 8 + r;
      pw[row * P_S + ln]      = (_Float16)p0[r];
      pw[row * P_S + 16 + ln] = (_Float16)p1[r];
    }
    asm volatile("s_wait_dscnt 0" ::: "memory");   // per-wave LDS RAW
    __builtin_amdgcn_wave_barrier();

    v16h ap;
    {
      union { v16h v; f32x4 f[2]; } u;
      const _Float16* pr = pw + (size_t)ln * P_S + lh * 8;
      u.f[0] = *(const f32x4*)(pr);
      u.f[1] = *(const f32x4*)(pr + 16);
      ap = u.v;
    }

    // ---- acc(16x256) += P(16x32) @ V(32x256), 16 column tiles ----
    #pragma unroll
    for (int t = 0; t < 16; ++t) {
      union { v16h v; f32x4 f[2]; } u;
      const _Float16* vr = vt + (size_t)(t * 16 + ln) * VT_S + lh * 8;
      u.f[0] = *(const f32x4*)(vr);
      u.f[1] = *(const f32x4*)(vr + 16);
      acc[t] = WMMA_F32_F16(ap, u.v, acc[t]);
    }
  }

  // ---- normalize and reduce across 24 heads ----
  float inv[8];
  #pragma unroll
  for (int r = 0; r < 8; ++r) inv[r] = 1.0f / s8[r];
  #pragma unroll
  for (int t = 0; t < 16; ++t)
    #pragma unroll
    for (int r = 0; r < 8; ++r) {
      const int row = i0 + lh * 8 + r;
      const int col = t * 16 + ln;
      atomicAdd(&out[(size_t)row * DDIM + col], acc[t][r] * inv[r]);
    }
}

// ---------------- host side ----------------
extern "C" void kernel_launch(void* const* d_in, const int* in_sizes, int n_in,
                              void* d_out, int out_size, void* d_ws, size_t ws_size,
                              hipStream_t stream) {
  (void)in_sizes; (void)n_in; (void)ws_size;
  const float* nodes   = (const float*)d_in[0];
  const float* aux     = (const float*)d_in[1];
  const float* rot     = (const float*)d_in[2];
  const float* W_nodes = (const float*)d_in[3];
  const float* b_nodes = (const float*)d_in[4];
  const float* W_aux   = (const float*)d_in[5];
  const float* b_aux   = (const float*)d_in[6];
  const float* W_rot   = (const float*)d_in[7];
  const float* W_val   = (const float*)d_in[8];
  const float* b_val   = (const float*)d_in[9];

  char* ws = (char*)d_ws;
  size_t off = 0;
  auto carve = [&](size_t elems) -> _Float16* {
    _Float16* p = (_Float16*)(ws + off);
    off += (elems * sizeof(_Float16) + 255) & ~(size_t)255;
    return p;
  };
  _Float16* nodes_h = carve((size_t)SEQ * 256);
  _Float16* aux_h   = carve((size_t)SEQ * 64);
  _Float16* rot_h   = carve((size_t)SEQ * 4);
  _Float16* Wn_h    = carve((size_t)1024 * 256);
  _Float16* Wa_h    = carve((size_t)1024 * 64);
  _Float16* Wr_h    = carve((size_t)1024 * 4);
  _Float16* Wv_h    = carve((size_t)6144 * 256);
  _Float16* val_h   = carve((size_t)SEQ * VLD);
  _Float16* k_h     = carve((size_t)SEQ * KQLD);
  _Float16* q_h     = carve((size_t)SEQ * KQLD);

  auto cvt = [&](const float* s, _Float16* d, int n) {
    cvt_f32_to_f16<<<(n + 255) / 256, 256, 0, stream>>>(s, d, n);
  };
  cvt(nodes,   nodes_h, SEQ * 256);
  cvt(aux,     aux_h,   SEQ * 64);
  cvt(rot,     rot_h,   SEQ * 4);
  cvt(W_nodes, Wn_h,    1024 * 256);
  cvt(W_aux,   Wa_h,    1024 * 64);
  cvt(W_rot,   Wr_h,    1024 * 4);
  cvt(W_val,   Wv_h,    6144 * 256);

  // val projection: (2048 x 256) @ (6144 x 256)^T + b_val -> (2048, 24, 256) f16
  wmma_proj_gemm<<<dim3(6144 / 16, SEQ / 16), 32, 0, stream>>>(
      nodes_h, Wv_h, b_val, val_h, VLD, nullptr, nullptr, 0, 256);
  // k/q projections, scattered into concatenated (S, 24, 64) layout
  wmma_proj_gemm<<<dim3(1024 / 16, SEQ / 16), 32, 0, stream>>>(
      nodes_h, Wn_h, b_nodes, nullptr, 0, k_h, q_h, 0, 256);
  wmma_proj_gemm<<<dim3(1024 / 16, SEQ / 16), 32, 0, stream>>>(
      aux_h, Wa_h, b_aux, nullptr, 0, k_h, q_h, 8, 64);
  wmma_proj_gemm<<<dim3(1024 / 16, SEQ / 16), 32, 0, stream>>>(
      rot_h, Wr_h, nullptr, nullptr, 0, k_h, q_h, 16, 4);

  hipMemsetAsync(d_out, 0, (size_t)out_size * sizeof(float), stream);

  // 24 heads x 32 i-groups; atomic head-reduction into d_out
  flash_attn_wmma<<<dim3(SEQ / 64, NH24), 128, 0, stream>>>(
      k_h, q_h, val_h, (float*)d_out);
}
